// SequenceModel_33225867002530
// MI455X (gfx1250) — matrix operand order repaired and checked by reading, not compiled, not run on hardware
//
#include <hip/hip_runtime.h>
#include <hip/hip_bf16.h>
#include <math.h>

// ---------------- problem constants ----------------
#define B_ 4
#define S_ 2048
#define D_ 512
#define H_ 8
#define DH_ 64
#define NL_ 6
#define DFF_ 2048
#define NP_C 64
#define NT_C 6001
#define NF_C 256
#define DP_C 64
#define DT_C 192
#define DF_C 256
#define M_ (B_ * S_)   // 8192

typedef __bf16 bf16;
typedef __attribute__((ext_vector_type(16))) __bf16 bf16x16;
typedef __attribute__((ext_vector_type(8)))  float  f32x8;

__device__ __forceinline__ f32x8 zero_f32x8() {
  f32x8 z;
#pragma unroll
  for (int i = 0; i < 8; ++i) z[i] = 0.0f;
  return z;
}

// ---------------- fp32 -> bf16 convert ----------------
__global__ void cvt_f32_bf16_kernel(const float* __restrict__ src,
                                    bf16* __restrict__ dst, long n) {
  long i = (long)blockIdx.x * 256 + threadIdx.x;
  if (i < n) dst[i] = (bf16)src[i];
}

// ---------------- embedding + sinusoidal PE ----------------
__global__ void embed_kernel(const int* __restrict__ part,
                             const int* __restrict__ dt,
                             const int* __restrict__ f0,
                             const float* __restrict__ tb,
                             const float* __restrict__ pe,
                             const float* __restrict__ te,
                             const float* __restrict__ fe,
                             float* __restrict__ x,
                             bf16* __restrict__ xb) {
  long idx = (long)blockIdx.x * 256 + threadIdx.x;
  if (idx >= (long)M_ * D_) return;
  int m = (int)(idx / D_);
  int d = (int)(idx % D_);
  float e;
  if (d < DP_C)              e = pe[(size_t)part[m] * DP_C + d];
  else if (d < DP_C + DT_C)  e = te[(size_t)dt[m]   * DT_C + (d - DP_C)];
  else                       e = fe[(size_t)f0[m]   * DF_C + (d - DP_C - DT_C)];
  int i = d >> 1;
  float wk  = __powf(10000.0f, -(float)i / (float)D_) * ((float)D_ / 36000.0f);
  float ang = tb[m] * wk;
  float pev = (d & 1) ? __cosf(ang) : __sinf(ang);
  float v = e + pev;
  x[idx]  = v;
  xb[idx] = (bf16)v;
}

// ---------------- generic bf16 WMMA GEMM: C[M,N] = A[M,K] * W[N,K]^T + bias ----------------
// Block: 256 thr = 8 waves arranged 4(M) x 2(N); block tile 128x128; wave tile 32x64.
// Branch-free inner loop: W row pointers are clamped to N-1 once, outside the K-loop;
// out-of-range columns are computed but never stored.
__global__ __launch_bounds__(256) void gemm_bf16_kernel(
    const bf16* __restrict__ A, int lda,
    const bf16* __restrict__ W, int ldw,
    const float* __restrict__ bias,
    float* __restrict__ Cf, bf16* __restrict__ Cb, int ldc,
    int M, int N, int K, int relu)
{
  int wave = threadIdx.x >> 5;
  int lane = threadIdx.x & 31;
  int wr = wave & 3, wc = wave >> 2;
  int m0 = blockIdx.y * 128 + wr * 32;
  int n0 = blockIdx.x * 128 + wc * 64;
  if (n0 >= N) return;                 // wave-uniform early exit (head GEMMs)
  int lm = lane & 15, half = lane >> 4;

  f32x8 acc[2][4];
#pragma unroll
  for (int u = 0; u < 2; ++u)
#pragma unroll
    for (int t = 0; t < 4; ++t) acc[u][t] = zero_f32x8();

  const bf16* arow0 = A + (size_t)(m0 + lm) * lda;
  const bf16* arow1 = A + (size_t)(m0 + 16 + lm) * lda;
  const bf16* wbase[4];
#pragma unroll
  for (int t = 0; t < 4; ++t) {
    int n = n0 + t * 16 + lm;
    int nc = (n < N) ? n : (N - 1);    // clamp -> loads stay in bounds, no divergence
    wbase[t] = W + (size_t)nc * ldw + 16 * half;
  }
  const int aoff0 = 16 * half;         // A-fragment: two contiguous 8-elem K runs
  const int aoff1 = 16 * half + 8;

  for (int k0 = 0; k0 < K; k0 += 32) {
    __builtin_prefetch((const void*)(arow0 + k0 + 64), 0, 1);
    __builtin_prefetch((const void*)(arow1 + k0 + 64), 0, 1);
    bf16x16 af0, af1;
#pragma unroll
    for (int e = 0; e < 8; ++e) {
      af0[e]     = arow0[k0 + aoff0 + e];
      af0[e + 8] = arow0[k0 + aoff1 + e];
      af1[e]     = arow1[k0 + aoff0 + e];
      af1[e + 8] = arow1[k0 + aoff1 + e];
    }
#pragma unroll
    for (int t = 0; t < 4; ++t) {
      bf16x16 bfg;
#pragma unroll
      for (int e = 0; e < 16; ++e) bfg[e] = wbase[t][k0 + e];
      acc[0][t] = __builtin_amdgcn_wmma_f32_16x16x32_bf16(
          false, af0, false, bfg, (short)0, acc[0][t], false, false);
      acc[1][t] = __builtin_amdgcn_wmma_f32_16x16x32_bf16(
          false, af1, false, bfg, (short)0, acc[1][t], false, false);
    }
  }

#pragma unroll
  for (int t = 0; t < 4; ++t) {
    int n = n0 + t * 16 + lm;
    if (n >= N) continue;
    float bv = bias ? bias[n] : 0.0f;
#pragma unroll
    for (int u = 0; u < 2; ++u) {
#pragma unroll
      for (int r = 0; r < 8; ++r) {
        int m = m0 + u * 16 + r + 8 * half;
        float v = acc[u][t][r] + bv;
        if (relu) v = fmaxf(v, 0.0f);
        size_t o = (size_t)m * ldc + n;
        if (Cf) Cf[o] = v;
        if (Cb) Cb[o] = (bf16)v;
      }
    }
  }
}

// ---------------- qkv fp32 [M,3D] -> q/k bf16 [BH,S,DH], v^T bf16 [BH,DH,S] ----------------
__global__ void qkv_pack_kernel(const float* __restrict__ qkv,
                                bf16* __restrict__ qb,
                                bf16* __restrict__ kb,
                                bf16* __restrict__ vT) {
  long idx = (long)blockIdx.x * 256 + threadIdx.x;
  if (idx >= (long)M_ * 3 * D_) return;
  int m = (int)(idx / (3 * D_));
  int n = (int)(idx % (3 * D_));
  int which = n / D_;
  int dfull = n % D_;
  int h = dfull / DH_, dh = dfull % DH_;
  int b = m / S_, s = m % S_;
  int bh = b * H_ + h;
  bf16 v = (bf16)qkv[idx];
  if (which == 0)      qb[((size_t)bh * S_ + s) * DH_ + dh] = v;
  else if (which == 1) kb[((size_t)bh * S_ + s) * DH_ + dh] = v;
  else                 vT[((size_t)bh * DH_ + dh) * S_ + s] = v;
}

// ---------------- flash attention (causal), 1 wave per (bh, 16-query tile) ----------------
__global__ __launch_bounds__(32) void flash_attn_kernel(
    const bf16* __restrict__ qb, const bf16* __restrict__ kb,
    const bf16* __restrict__ vT, float* __restrict__ o)
{
  __shared__ bf16 pLds[16 * 32];
  int bh = blockIdx.y;
  int q0 = blockIdx.x * 16;
  int lane = threadIdx.x & 31;
  int lm = lane & 15, half = lane >> 4;
  const float scale = 0.125f;

  // Q fragments over dh: two 16x32 A-tiles
  bf16x16 qa[2];
  {
    const bf16* qrow = qb + ((size_t)bh * S_ + (q0 + lm)) * DH_;
#pragma unroll
    for (int c = 0; c < 2; ++c) {
#pragma unroll
      for (int e = 0; e < 8; ++e)  qa[c][e] = qrow[c * 32 + 16 * half + e];
#pragma unroll
      for (int e = 8; e < 16; ++e) qa[c][e] = qrow[c * 32 + 16 * half + e];
    }
  }

  f32x8 oacc[4];
#pragma unroll
  for (int t = 0; t < 4; ++t) oacc[t] = zero_f32x8();
  float mrow[8], lrow[8];
#pragma unroll
  for (int r = 0; r < 8; ++r) { mrow[r] = -1e30f; lrow[r] = 0.0f; }

  int kend = q0 + 16;
  for (int kt = 0; kt < kend; kt += 32) {
    // scores for two 16-key tiles: C = Q (16xDH) * K^T (DHx16)
    f32x8 s0 = zero_f32x8(), s1 = zero_f32x8();
#pragma unroll
    for (int kt2 = 0; kt2 < 2; ++kt2) {
      f32x8 sc = zero_f32x8();
      int key = kt + kt2 * 16 + lm;
#pragma unroll
      for (int c = 0; c < 2; ++c) {
        bf16x16 kf;
        const bf16* krow = kb + ((size_t)bh * S_ + key) * DH_ + c * 32 + 16 * half;
#pragma unroll
        for (int e = 0; e < 16; ++e) kf[e] = krow[e];
        sc = __builtin_amdgcn_wmma_f32_16x16x32_bf16(
            false, qa[c], false, kf, (short)0, sc, false, false);
      }
      if (kt2 == 0) s0 = sc; else s1 = sc;
    }

    // online softmax per row (rows r+8*half live across the 16 lanes of each half)
#pragma unroll
    for (int r = 0; r < 8; ++r) {
      int q = q0 + r + 8 * half;
      float v0 = s0[r] * scale + (((kt + lm) > q)      ? -1e9f : 0.0f);
      float v1 = s1[r] * scale + (((kt + 16 + lm) > q) ? -1e9f : 0.0f);
      float cmax = fmaxf(v0, v1);
#pragma unroll
      for (int off = 8; off; off >>= 1) cmax = fmaxf(cmax, __shfl_xor(cmax, off, 32));
      float mnew = fmaxf(mrow[r], cmax);
      float p0 = __expf(v0 - mnew);
      float p1 = __expf(v1 - mnew);
      float rs = p0 + p1;
#pragma unroll
      for (int off = 8; off; off >>= 1) rs += __shfl_xor(rs, off, 32);
      float alpha = __expf(mrow[r] - mnew);
      lrow[r] = lrow[r] * alpha + rs;
      mrow[r] = mnew;
#pragma unroll
      for (int t = 0; t < 4; ++t) oacc[t][r] *= alpha;
      pLds[(r + 8 * half) * 32 + lm]      = (bf16)p0;
      pLds[(r + 8 * half) * 32 + 16 + lm] = (bf16)p1;
    }
    __syncthreads();

    // P as A-fragment (16 rows x 32 keys) from LDS
    bf16x16 pa;
#pragma unroll
    for (int e = 0; e < 8; ++e)  pa[e] = pLds[lm * 32 + 16 * half + e];
#pragma unroll
    for (int e = 8; e < 16; ++e) pa[e] = pLds[lm * 32 + 16 * half + e];
    __syncthreads();

    // O += P * V  (V^T stored [bh, dh, s] -> contiguous B-fragment runs)
#pragma unroll
    for (int t = 0; t < 4; ++t) {
      bf16x16 vf;
      const bf16* vrow = vT + ((size_t)bh * DH_ + t * 16 + lm) * S_ + kt + 16 * half;
#pragma unroll
      for (int e = 0; e < 16; ++e) vf[e] = vrow[e];
      oacc[t] = __builtin_amdgcn_wmma_f32_16x16x32_bf16(
          false, pa, false, vf, (short)0, oacc[t], false, false);
    }
  }

  int b = bh / H_, h = bh % H_;
#pragma unroll
  for (int t = 0; t < 4; ++t) {
    int dh = t * 16 + lm;
#pragma unroll
    for (int r = 0; r < 8; ++r) {
      int q = q0 + r + 8 * half;
      float v = oacc[t][r] / lrow[r];
      o[((size_t)(b * S_ + q)) * D_ + h * DH_ + dh] = v;
    }
  }
}

// ---------------- fused residual add + LayerNorm (one block per row) ----------------
__global__ __launch_bounds__(256) void add_ln_kernel(
    float* __restrict__ x, const float* __restrict__ res,
    const float* __restrict__ sc, const float* __restrict__ bi,
    bf16* __restrict__ xb)
{
  __shared__ float red[2][8];
  int m = blockIdx.x;
  int tid = threadIdx.x;
  size_t base = (size_t)m * D_;
  float v0 = x[base + tid]       + res[base + tid];
  float v1 = x[base + 256 + tid] + res[base + 256 + tid];
  float sum = v0 + v1, sq = v0 * v0 + v1 * v1;
#pragma unroll
  for (int off = 16; off; off >>= 1) {
    sum += __shfl_xor(sum, off, 32);
    sq  += __shfl_xor(sq,  off, 32);
  }
  int w = tid >> 5;
  if ((tid & 31) == 0) { red[0][w] = sum; red[1][w] = sq; }
  __syncthreads();
  sum = 0.0f; sq = 0.0f;
#pragma unroll
  for (int i = 0; i < 8; ++i) { sum += red[0][i]; sq += red[1][i]; }
  float mean = sum * (1.0f / D_);
  float var  = sq * (1.0f / D_) - mean * mean;
  float rstd = rsqrtf(var + 1e-5f);
  float o0 = (v0 - mean) * rstd * sc[tid]       + bi[tid];
  float o1 = (v1 - mean) * rstd * sc[256 + tid] + bi[256 + tid];
  x[base + tid]        = o0;
  x[base + 256 + tid]  = o1;
  xb[base + tid]       = (bf16)o0;
  xb[base + 256 + tid] = (bf16)o1;
}

// ---------------- host-side orchestration ----------------
static inline size_t align256(size_t x) { return (x + 255) & ~(size_t)255; }

extern "C" void kernel_launch(void* const* d_in, const int* in_sizes, int n_in,
                              void* d_out, int out_size, void* d_ws, size_t ws_size,
                              hipStream_t stream) {
  (void)in_sizes; (void)n_in; (void)out_size; (void)ws_size;
  const int*   participant = (const int*)d_in[0];
  const int*   delta_time  = (const int*)d_in[1];
  const int*   f0idx       = (const int*)d_in[2];
  const float* time_back   = (const float*)d_in[3];
  const float* part_emb    = (const float*)d_in[4];
  const float* time_emb    = (const float*)d_in[5];
  const float* f0_emb      = (const float*)d_in[6];
  const float* Wqkv        = (const float*)d_in[7];
  const float* bqkv        = (const float*)d_in[8];
  const float* Wo          = (const float*)d_in[9];
  const float* bo          = (const float*)d_in[10];
  const float* W1          = (const float*)d_in[11];
  const float* b1          = (const float*)d_in[12];
  const float* W2          = (const float*)d_in[13];
  const float* b2          = (const float*)d_in[14];
  const float* ln1_s       = (const float*)d_in[15];
  const float* ln1_b       = (const float*)d_in[16];
  const float* ln2_s       = (const float*)d_in[17];
  const float* ln2_b       = (const float*)d_in[18];
  float* out = (float*)d_out;

  // workspace layout
  char* ws = (char*)d_ws;
  size_t off = 0;
  float* x    = (float*)(ws + off); off = align256(off + (size_t)M_ * D_ * 4);
  bf16*  xb   = (bf16*)(ws + off);  off = align256(off + (size_t)M_ * D_ * 2);
  bf16*  qb   = (bf16*)(ws + off);  off = align256(off + (size_t)M_ * D_ * 2);
  bf16*  kb   = (bf16*)(ws + off);  off = align256(off + (size_t)M_ * D_ * 2);
  bf16*  vT   = (bf16*)(ws + off);  off = align256(off + (size_t)M_ * D_ * 2);
  float* big  = (float*)(ws + off); off = align256(off + (size_t)M_ * 3 * D_ * 4); // qkv f32 / attn-o f32
  bf16*  ob   = (bf16*)(ws + off);  off = align256(off + (size_t)M_ * D_ * 2);
  float* tmp2 = (float*)(ws + off); off = align256(off + (size_t)M_ * D_ * 4);
  bf16*  h1b  = (bf16*)(ws + off);  off = align256(off + (size_t)M_ * DFF_ * 2);
  bf16*  wqkvB = (bf16*)(ws + off); off = align256(off + (size_t)NL_ * 3 * D_ * D_ * 2);
  bf16*  woB   = (bf16*)(ws + off); off = align256(off + (size_t)NL_ * D_ * D_ * 2);
  bf16*  w1B   = (bf16*)(ws + off); off = align256(off + (size_t)NL_ * DFF_ * D_ * 2);
  bf16*  w2B   = (bf16*)(ws + off); off = align256(off + (size_t)NL_ * D_ * DFF_ * 2);
  bf16*  peB   = (bf16*)(ws + off); off = align256(off + (size_t)NP_C * DP_C * 2);
  bf16*  teB   = (bf16*)(ws + off); off = align256(off + (size_t)NT_C * DT_C * 2);
  bf16*  feB   = (bf16*)(ws + off); off = align256(off + (size_t)NF_C * DF_C * 2);

  auto cvt = [&](const float* s, bf16* d, long n) {
    cvt_f32_bf16_kernel<<<(unsigned)((n + 255) / 256), 256, 0, stream>>>(s, d, n);
  };
  cvt(Wqkv, wqkvB, (long)NL_ * 3 * D_ * D_);
  cvt(Wo,   woB,   (long)NL_ * D_ * D_);
  cvt(W1,   w1B,   (long)NL_ * DFF_ * D_);
  cvt(W2,   w2B,   (long)NL_ * D_ * DFF_);
  cvt(part_emb, peB, (long)NP_C * DP_C);
  cvt(time_emb, teB, (long)NT_C * DT_C);
  cvt(f0_emb,   feB, (long)NF_C * DF_C);

  {
    long n = (long)M_ * D_;
    embed_kernel<<<(unsigned)((n + 255) / 256), 256, 0, stream>>>(
        participant, delta_time, f0idx, time_back, part_emb, time_emb, f0_emb, x, xb);
  }

  auto gemm = [&](const bf16* A, int lda, const bf16* W, int ldw, const float* bias,
                  float* Cf, bf16* Cb, int ldc, int M, int N, int K, int relu) {
    dim3 grid((N + 127) / 128, M / 128);
    gemm_bf16_kernel<<<grid, 256, 0, stream>>>(A, lda, W, ldw, bias, Cf, Cb, ldc, M, N, K, relu);
  };

  for (int l = 0; l < NL_; ++l) {
    // QKV projection
    gemm(xb, D_, wqkvB + (size_t)l * 3 * D_ * D_, D_, bqkv + (size_t)l * 3 * D_,
         big, nullptr, 3 * D_, M_, 3 * D_, D_, 0);
    {
      long n = (long)M_ * 3 * D_;
      qkv_pack_kernel<<<(unsigned)((n + 255) / 256), 256, 0, stream>>>(big, qb, kb, vT);
    }
    // causal flash attention -> big (reused as [M,D] fp32)
    {
      dim3 grid(S_ / 16, B_ * H_);
      flash_attn_kernel<<<grid, 32, 0, stream>>>(qb, kb, vT, big);
    }
    cvt(big, ob, (long)M_ * D_);
    // output projection
    gemm(ob, D_, woB + (size_t)l * D_ * D_, D_, bo + (size_t)l * D_,
         tmp2, nullptr, D_, M_, D_, D_, 0);
    add_ln_kernel<<<M_, 256, 0, stream>>>(x, tmp2, ln1_s + (size_t)l * D_, ln1_b + (size_t)l * D_, xb);
    // FFN
    gemm(xb, D_, w1B + (size_t)l * DFF_ * D_, D_, b1 + (size_t)l * DFF_,
         nullptr, h1b, DFF_, M_, DFF_, D_, 1);
    gemm(h1b, DFF_, w2B + (size_t)l * D_ * DFF_, DFF_, b2 + (size_t)l * D_,
         tmp2, nullptr, D_, M_, D_, DFF_, 0);
    add_ln_kernel<<<M_, 256, 0, stream>>>(x, tmp2, ln2_s + (size_t)l * D_, ln2_b + (size_t)l * D_, xb);
  }

  // output heads (column slices of x)
  float* out_cat  = out;
  float* out_time = out + (size_t)M_ * NP_C;
  float* out_f0   = out + (size_t)M_ * NP_C + (size_t)M_ * NT_C;
  gemm(xb,                 D_, peB, DP_C, nullptr, out_cat,  nullptr, NP_C, M_, NP_C, DP_C, 0);
  gemm(xb + DP_C,          D_, teB, DT_C, nullptr, out_time, nullptr, NT_C, M_, NT_C, DT_C, 0);
  gemm(xb + DP_C + DT_C,   D_, feB, DF_C, nullptr, out_f0,   nullptr, NF_C, M_, NF_C, DF_C, 0);
}
